// MiniGPT_20340965114366
// MI455X (gfx1250) — matrix operand order repaired and checked by reading, not compiled, not run on hardware
//
#include <hip/hip_runtime.h>
#include <hip/hip_bf16.h>
#include <math.h>

// ---------------------------------------------------------------------------
// MiniGPT forward on MI455X (gfx1250): wave32, WMMA bf16 16x16x32 (f32 accum),
// TDM tensor_load_to_lds for GEMM B-tiles where the builtin is available.
// ---------------------------------------------------------------------------

#define Bq 4
#define Tq 1024
#define Lq 6
#define Hq 6
#define Cq 384
#define Dq 64
#define Vq 50257
#define ROWS (Bq * Tq)          // 4096
#define C3 (3 * Cq)             // 1152
#define C4 (4 * Cq)             // 1536

typedef __attribute__((ext_vector_type(16))) __bf16 bf16x16;
typedef __attribute__((ext_vector_type(8)))  __bf16 bf16x8;
typedef __attribute__((ext_vector_type(8)))  float  f32x8;
typedef __attribute__((ext_vector_type(4)))  float  f32x4;
typedef __attribute__((ext_vector_type(4)))  unsigned int u32x4;
typedef __attribute__((ext_vector_type(4)))  int i32x4;
typedef __attribute__((ext_vector_type(8)))  int i32x8;

#if defined(__gfx1250__) && __has_builtin(__builtin_amdgcn_tensor_load_to_lds)
#define USE_TDM 1
#endif

__device__ __forceinline__ f32x8 wmma_bf16(bf16x16 a, bf16x16 b, f32x8 c) {
  // emits v_wmma_f32_16x16x32_bf16
  return __builtin_amdgcn_wmma_f32_16x16x32_bf16(
      /*neg_a=*/false, a, /*neg_b=*/false, b,
      /*c_mod=*/(short)0, c, /*reuse_a=*/false, /*reuse_b=*/false);
}

__device__ __forceinline__ bf16x16 ld_frag_pair(const __bf16* lo, const __bf16* hi) {
  bf16x8 a = *reinterpret_cast<const bf16x8*>(lo);
  bf16x8 b = *reinterpret_cast<const bf16x8*>(hi);
  bf16x16 r;
#pragma unroll
  for (int i = 0; i < 8; ++i) { r[i] = a[i]; r[i + 8] = b[i]; }
  return r;
}

__device__ __forceinline__ bf16x16 ld_frag_col(const __bf16* p, int ld) {
  bf16x16 r;
#pragma unroll
  for (int i = 0; i < 8; ++i) { r[i] = p[i * ld]; r[i + 8] = p[(i + 16) * ld]; }
  return r;
}

__device__ __forceinline__ bf16x16 cvt_col(const float* p, int ld) {
  bf16x16 r;
#pragma unroll
  for (int i = 0; i < 8; ++i) { r[i] = (__bf16)p[i * ld]; r[i + 8] = (__bf16)p[(i + 16) * ld]; }
  return r;
}

// 16 contiguous fp32 (16B aligned) -> bf16x16 fragment (element i -> slot i)
__device__ __forceinline__ bf16x16 cvt16(const float* p) {
  const f32x4* q = (const f32x4*)p;
  f32x4 a = q[0], b = q[1], c = q[2], d = q[3];
  bf16x16 r;
#pragma unroll
  for (int i = 0; i < 4; ++i) {
    r[i]      = (__bf16)a[i];
    r[i + 4]  = (__bf16)b[i];
    r[i + 8]  = (__bf16)c[i];
    r[i + 12] = (__bf16)d[i];
  }
  return r;
}

#ifdef USE_TDM
// DMA a 32x128 fp32 sub-tile (row stride N elements) of B into LDS at lds_off.
// LDS layout after pad: 32 rows x (128 + 4) dwords  (pad 4 dw per 128 dw).
__device__ __forceinline__ void tdm_load_b_tile(const float* gsrc, unsigned lds_off,
                                                int N, int K) {
  unsigned long long ga = (unsigned long long)(size_t)(const void*)gsrc;
  u32x4 g0 = { 1u,                                     // count=1, is_restore=0
               lds_off,                                // lds_addr (bytes)
               (unsigned)ga,
               (unsigned)(ga >> 32) | (2u << 30) };    // addr[56:32] | type=2
  unsigned td0 = (unsigned)N, td1 = (unsigned)K, st0 = (unsigned)N;
  i32x8 g1 = {
    (int)((2u << 16) | (1u << 20) | (6u << 22) | (3u << 25)), // dsz=4B,pad_en,int=128dw,amt=4dw
    (int)((td0 & 0xFFFFu) << 16),                     // tensor_dim0 lo16 (hi of dw1)
    (int)((td0 >> 16) | ((td1 & 0xFFFFu) << 16)),     // td0 hi16 | td1 lo16
    (int)((td1 >> 16) | (128u << 16)),                // td1 hi16 | tile_dim0=128
    (int)32,                                          // tile_dim1=32, tile_dim2=0
    (int)st0,                                         // tensor_dim0_stride lo32
    0, 0 };
  i32x4 gz = { 0, 0, 0, 0 };
#if __clang_major__ >= 23
  i32x8 gz8 = { 0, 0, 0, 0, 0, 0, 0, 0 };
  __builtin_amdgcn_tensor_load_to_lds(g0, g1, gz, gz, gz8, 0);
#else
  __builtin_amdgcn_tensor_load_to_lds(g0, g1, gz, gz, 0);
#endif
}
#endif

// ---------------------------------------------------------------------------
// Embedding: x[row, c] = wte[idx[row], c] + wpe[row % T, c]
// ---------------------------------------------------------------------------
__global__ __launch_bounds__(Cq) void embed_kernel(const int* __restrict__ idx,
                                                   const float* __restrict__ wte,
                                                   const float* __restrict__ wpe,
                                                   float* __restrict__ x) {
  int row = blockIdx.x;
  int c = threadIdx.x;
  int t = row & (Tq - 1);
  int tok = idx[row];
  x[(size_t)row * Cq + c] = wte[(size_t)tok * Cq + c] + wpe[(size_t)t * Cq + c];
}

// ---------------------------------------------------------------------------
// LayerNorm over C=384, one 128-thread workgroup per row, fp32 in/out.
// ---------------------------------------------------------------------------
__global__ __launch_bounds__(128) void ln_kernel(const float* __restrict__ in,
                                                 const float* __restrict__ w,
                                                 const float* __restrict__ b,
                                                 float* __restrict__ out) {
  __shared__ float red[128];
  int row = blockIdx.x;
  int tid = threadIdx.x;
  const float* rp = in + (size_t)row * Cq;

  float s = 0.f;
  for (int c = tid; c < Cq; c += 128) s += rp[c];
  red[tid] = s; __syncthreads();
  for (int o = 64; o > 0; o >>= 1) { if (tid < o) red[tid] += red[tid + o]; __syncthreads(); }
  float mu = red[0] * (1.0f / Cq);
  __syncthreads();

  float v = 0.f;
  for (int c = tid; c < Cq; c += 128) { float d = rp[c] - mu; v += d * d; }
  red[tid] = v; __syncthreads();
  for (int o = 64; o > 0; o >>= 1) { if (tid < o) red[tid] += red[tid + o]; __syncthreads(); }
  float rstd = rsqrtf(red[0] * (1.0f / Cq) + 1e-5f);

  float* op = out + (size_t)row * Cq;
  for (int c = tid; c < Cq; c += 128)
    op[c] = (rp[c] - mu) * rstd * w[c] + b[c];
}

// ---------------------------------------------------------------------------
// GEMM: out[M,N] (ld = Nreal) = A[M,K] (fp32) * B + bias, bf16 WMMA inner.
//   btrans == 0: B is [K,N] row-major fp32 (weights); B tile via TDM if avail.
//   btrans == 1: B[k,n] = Bmat[n*K + k]  (tied lm-head: Bmat = wte[V,C])
//   ep: 0 = bias, 1 = bias + residual-add into out, 2 = bias + exact GELU,
//       3 = no bias, N-guarded store (logits)
// Tile 64x128, BK=32, 256 threads = 8 waves, each wave -> 16x64 of D (4 frags).
// ---------------------------------------------------------------------------
__global__ __launch_bounds__(256) void gemm_bf16_kernel(
    const float* __restrict__ A, const float* __restrict__ Bmat,
    const float* __restrict__ bias, float* __restrict__ out,
    int M, int N, int K, int Nreal, int ep, int btrans) {
  __shared__ __align__(16) __bf16 As[64][48];    // A tile bf16: 32 cols + pad
  __shared__ __align__(16) __bf16 Bs[32][136];   // B tile bf16 (fallback/btrans)
#ifdef USE_TDM
  __shared__ __align__(16) float Bs32[32][132];  // B tile fp32 via TDM (4dw row pad)
#endif

  int n0 = blockIdx.x * 128;
  int m0 = blockIdx.y * 64;
  int tid = threadIdx.x;
  int w = tid >> 5, lane = tid & 31;
  int mw = (w & 3) * 16;            // wave row subtile
  int nw = (w >> 2) * 64;           // wave col base (4 frags of 16)
  int ml = lane & 15, hi = lane >> 4, kb8 = hi * 8;

  f32x8 acc[4] = {};

  for (int k0 = 0; k0 < K; k0 += 32) {
    __syncthreads();
    { // stage A tile 64x32 fp32 -> bf16 (registers -> LDS)
      int row = tid >> 2, col = (tid & 3) * 8;
      const float* src = A + (size_t)(m0 + row) * K + k0 + col;
      if (k0 + 32 < K) __builtin_prefetch(src + 32, 0, 1);   // global_prefetch_b8
      bf16x8 t;
#pragma unroll
      for (int i = 0; i < 8; ++i) t[i] = (__bf16)src[i];
      *reinterpret_cast<bf16x8*>(&As[row][col]) = t;
    }

    bool tdm_b = false;
#ifdef USE_TDM
    if (!btrans) {
      tdm_b = true;
      if (w == 0) {   // wave-uniform: only wave 0 issues the DMA (TDM ignores EXEC)
        tdm_load_b_tile(Bmat + (size_t)(k0)*N + n0,
                        (unsigned)(size_t)(const void*)&Bs32[0][0], N, K);
        __builtin_amdgcn_s_wait_tensorcnt(0);
      }
    }
#endif
    if (!tdm_b) {
      if (!btrans) { // stage B tile 32x128 fp32 -> bf16
        int k = tid >> 3, n = (tid & 7) * 16;
        const float* src = Bmat + (size_t)(k0 + k) * N + n0 + n;
        if (k0 + 32 < K) __builtin_prefetch(src + (size_t)32 * N, 0, 1);
        bf16x8 t0, t1;
#pragma unroll
        for (int i = 0; i < 8; ++i) { t0[i] = (__bf16)src[i]; t1[i] = (__bf16)src[i + 8]; }
        *reinterpret_cast<bf16x8*>(&Bs[k][n])     = t0;
        *reinterpret_cast<bf16x8*>(&Bs[k][n + 8]) = t1;
      } else {       // B[k][n] = Bmat[n*K + k], branch-free transpose staging
        int n = tid >> 1, k = (tid & 1) * 16;
        int nglob = n0 + n;
        int nsafe = (nglob < Nreal) ? nglob : (Nreal - 1);
        float zf = (nglob < Nreal) ? 1.0f : 0.0f;
        const float* src = Bmat + (size_t)nsafe * K + k0 + k;
#pragma unroll
        for (int i = 0; i < 16; ++i)
          Bs[k + i][n] = (__bf16)(src[i] * zf);
      }
    }
    __syncthreads();

    // fragments per ISA 7.12.2 layouts
    bf16x16 af = ld_frag_pair(&As[mw + ml][kb8], &As[mw + ml][kb8 + 16]);
#pragma unroll
    for (int fi = 0; fi < 4; ++fi) {
      bf16x16 bfr;
#ifdef USE_TDM
      if (tdm_b) bfr = cvt_col(&Bs32[kb8][nw + fi * 16 + ml], 132);
      else
#endif
        bfr = ld_frag_col(&Bs[kb8][nw + fi * 16 + ml], 136);
      acc[fi] = wmma_bf16(af, bfr, acc[fi]);
    }
  }

  // epilogue: lane holds (n = nw + fi*16 + ml), rows m = mw + v + hi*8
#pragma unroll
  for (int fi = 0; fi < 4; ++fi) {
    int n = n0 + nw + fi * 16 + ml;
    if (n < Nreal) {
      float bv = bias ? bias[n] : 0.0f;
#pragma unroll
      for (int v = 0; v < 8; ++v) {
        int mr = m0 + mw + v + hi * 8;
        size_t off = (size_t)mr * Nreal + n;
        float val = acc[fi][v] + bv;
        if (ep == 1)      out[off] += val;                                   // residual
        else if (ep == 2) out[off] = 0.5f * val * (1.0f + erff(val * 0.70710678118f));
        else              out[off] = val;                                    // 0 or 3
      }
    }
  }
}

// ---------------------------------------------------------------------------
// Causal flash attention, head_dim=64, 64 queries per WG, 4 waves x 16 rows.
// qkv fp32 [B*T, 3C]; y fp32 [B*T, C]. scale = 1/8.
// ---------------------------------------------------------------------------
__global__ __launch_bounds__(128) void attn_kernel(const float* __restrict__ qkv,
                                                   float* __restrict__ y) {
  __shared__ __align__(16) __bf16 Kt[64][72];       // transposed K block: [d][j]
  __shared__ __align__(16) __bf16 Vs[64][72];       // V block: [j][d]
  __shared__ __align__(16) __bf16 Ps[4][16][72];    // per-wave P tile

  int bh = blockIdx.y;
  int b = bh / Hq, h = bh % Hq;
  int qb = blockIdx.x;
  int tid = threadIdx.x;
  int w = tid >> 5, lane = tid & 31;
  int ml = lane & 15, hi = lane >> 4, kb8 = hi * 8;
  int q0w = qb * 64 + w * 16;

  // Q fragments (16x64 over k=d, two 32-wide k-chunks), fp32 -> bf16
  const float* qrow = qkv + (size_t)(b * Tq + q0w + ml) * C3 + h * Dq;
  bf16x16 qf[2];
#pragma unroll
  for (int f = 0; f < 2; ++f) {
#pragma unroll
    for (int j = 0; j < 8; ++j) {
      qf[f][j]     = (__bf16)qrow[f * 32 + kb8 + j];
      qf[f][j + 8] = (__bf16)qrow[f * 32 + kb8 + 16 + j];
    }
  }

  f32x8 o[4] = {};
  float mrow[8], lrow[8];
#pragma unroll
  for (int v = 0; v < 8; ++v) { mrow[v] = -1e30f; lrow[v] = 0.f; }

  for (int kb = 0; kb <= qb; ++kb) {
    __syncthreads();
    for (int it = tid; it < 64 * 64; it += 128) {       // stage K (transposed) + V
      int j = it >> 6, d = it & 63;
      const float* base = qkv + (size_t)(b * Tq + kb * 64 + j) * C3 + h * Dq + d;
      Kt[d][j] = (__bf16)base[Cq];
      Vs[j][d] = (__bf16)base[2 * Cq];
    }
    __syncthreads();

    // S = Q * K^T  (4 col subtiles x 2 k-steps)
    f32x8 s[4];
#pragma unroll
    for (int ns = 0; ns < 4; ++ns) {
      f32x8 acc = {};
#pragma unroll
      for (int ks = 0; ks < 2; ++ks) {
        bf16x16 bf = ld_frag_col(&Kt[ks * 32 + kb8][ns * 16 + ml], 72);
        acc = wmma_bf16(qf[ks], bf, acc);
      }
      s[ns] = acc;
    }

    // scale + causal mask + online softmax (rows live across 16-lane halves)
#pragma unroll
    for (int ns = 0; ns < 4; ++ns) {
      int key = kb * 64 + ns * 16 + ml;
#pragma unroll
      for (int v = 0; v < 8; ++v) {
        int q = qb * 64 + w * 16 + hi * 8 + v;
        float val = s[ns][v] * 0.125f;
        s[ns][v] = (key <= q) ? val : -1e30f;
      }
    }
#pragma unroll
    for (int v = 0; v < 8; ++v) {
      float mx = fmaxf(fmaxf(s[0][v], s[1][v]), fmaxf(s[2][v], s[3][v]));
#pragma unroll
      for (int d = 1; d < 16; d <<= 1) mx = fmaxf(mx, __shfl_xor(mx, d, 32));
      float mnew = fmaxf(mrow[v], mx);
      float corr = __expf(mrow[v] - mnew);
      mrow[v] = mnew;
      lrow[v] *= corr;
#pragma unroll
      for (int df = 0; df < 4; ++df) o[df][v] *= corr;
      float psum = 0.f;
#pragma unroll
      for (int ns = 0; ns < 4; ++ns) {
        float p = __expf(s[ns][v] - mnew);
        s[ns][v] = p;
        psum += p;
      }
#pragma unroll
      for (int d = 1; d < 16; d <<= 1) psum += __shfl_xor(psum, d, 32);
      lrow[v] += psum;
    }

    // round-trip P through per-wave LDS to re-enter A-fragment layout
#pragma unroll
    for (int ns = 0; ns < 4; ++ns)
#pragma unroll
      for (int v = 0; v < 8; ++v)
        Ps[w][v + hi * 8][ns * 16 + ml] = (__bf16)s[ns][v];

    // O += P * V
#pragma unroll
    for (int ks2 = 0; ks2 < 2; ++ks2) {
      bf16x16 pa = ld_frag_pair(&Ps[w][ml][ks2 * 32 + kb8],
                                &Ps[w][ml][ks2 * 32 + kb8 + 16]);
#pragma unroll
      for (int df = 0; df < 4; ++df) {
        bf16x16 vf = ld_frag_col(&Vs[ks2 * 32 + kb8][df * 16 + ml], 72);
        o[df] = wmma_bf16(pa, vf, o[df]);
      }
    }
  }

  // write y = O / l
#pragma unroll
  for (int df = 0; df < 4; ++df)
#pragma unroll
    for (int v = 0; v < 8; ++v) {
      int q = q0w + v + hi * 8;
      y[(size_t)(b * Tq + q) * Cq + h * Dq + df * 16 + ml] = o[df][v] / lrow[v];
    }
}

// ---------------------------------------------------------------------------
// Host-side orchestration
// ---------------------------------------------------------------------------
static void launch_gemm(hipStream_t s, const float* A, const float* Bm,
                        const float* bias, float* out,
                        int M, int N, int K, int Nreal, int ep, int btrans) {
  dim3 grid((Nreal + 127) / 128, M / 64);
  gemm_bf16_kernel<<<grid, 256, 0, s>>>(A, Bm, bias, out, M, N, K, Nreal, ep, btrans);
}

extern "C" void kernel_launch(void* const* d_in, const int* in_sizes, int n_in,
                              void* d_out, int out_size, void* d_ws, size_t ws_size,
                              hipStream_t stream) {
  (void)in_sizes; (void)n_in; (void)out_size; (void)ws_size;
  const int*   idx     = (const int*)  d_in[0];
  const float* wte     = (const float*)d_in[1];
  const float* wpe     = (const float*)d_in[2];
  const float* ln1_w   = (const float*)d_in[3];
  const float* ln1_b   = (const float*)d_in[4];
  const float* attn_W  = (const float*)d_in[5];
  const float* attn_b  = (const float*)d_in[6];
  const float* aproj_W = (const float*)d_in[7];
  const float* aproj_b = (const float*)d_in[8];
  const float* ln2_w   = (const float*)d_in[9];
  const float* ln2_b   = (const float*)d_in[10];
  const float* fc_W    = (const float*)d_in[11];
  const float* fc_b    = (const float*)d_in[12];
  const float* mproj_W = (const float*)d_in[13];
  const float* mproj_b = (const float*)d_in[14];
  const float* lnf_w   = (const float*)d_in[15];
  const float* lnf_b   = (const float*)d_in[16];
  float* logits = (float*)d_out;

  // workspace carve (all fp32): x | h | y | qkv | hfc  (~63 MB)
  float* x   = (float*)d_ws;
  float* h   = x   + (size_t)ROWS * Cq;
  float* y   = h   + (size_t)ROWS * Cq;
  float* qkv = y   + (size_t)ROWS * Cq;
  float* hfc = qkv + (size_t)ROWS * C3;

  embed_kernel<<<ROWS, Cq, 0, stream>>>(idx, wte, wpe, x);

  for (int l = 0; l < Lq; ++l) {
    const float* aW = attn_W  + (size_t)l * Cq * C3;
    const float* ab = attn_b  + (size_t)l * C3;
    const float* pW = aproj_W + (size_t)l * Cq * Cq;
    const float* pb = aproj_b + (size_t)l * Cq;
    const float* fW = fc_W    + (size_t)l * Cq * C4;
    const float* fb = fc_b    + (size_t)l * C4;
    const float* mW = mproj_W + (size_t)l * C4 * Cq;
    const float* mb = mproj_b + (size_t)l * Cq;

    ln_kernel<<<ROWS, 128, 0, stream>>>(x, ln1_w + l * Cq, ln1_b + l * Cq, h);
    launch_gemm(stream, h, aW, ab, qkv, ROWS, C3, Cq, C3, /*ep=*/0, /*bt=*/0);
    attn_kernel<<<dim3(Tq / 64, Bq * Hq), 128, 0, stream>>>(qkv, y);
    launch_gemm(stream, y, pW, pb, x, ROWS, Cq, Cq, Cq, /*ep=*/1, /*bt=*/0);   // residual
    ln_kernel<<<ROWS, 128, 0, stream>>>(x, ln2_w + l * Cq, ln2_b + l * Cq, h);
    launch_gemm(stream, h, fW, fb, hfc, ROWS, C4, Cq, C4, /*ep=*/2, /*bt=*/0); // GELU
    launch_gemm(stream, hfc, mW, mb, x, ROWS, Cq, C4, Cq, /*ep=*/1, /*bt=*/0); // residual
  }

  ln_kernel<<<ROWS, 128, 0, stream>>>(x, lnf_w, lnf_b, h);
  // logits = h @ wte^T : B staged transposed from wte, N-guarded stores
  launch_gemm(stream, h, wte, nullptr, logits, ROWS, Vq, Cq, Vq, /*ep=*/3, /*bt=*/1);
}
